// BlockGatingUnit_71751723647425
// MI455X (gfx1250) — compile-verified, matrix operand order checked by reading, not run on hardware
//
#include <hip/hip_runtime.h>

typedef __attribute__((ext_vector_type(16))) _Float16 v16h;
typedef __attribute__((ext_vector_type(8)))  float    v8f;
typedef __attribute__((ext_vector_type(4)))  float    v4f;

#define XB  16777216u   // x elems per batch  (256*256*256)
#define VB   8388608u   // v/u/out elems per batch (128*256*256)
#define EPS 1e-5f

// workspace byte offsets
#define BHI_OFF 256u                       // 16 ntiles * 8 ksteps * 32 lanes * 32B = 128KB
#define BLO_OFF (256u + 131072u)

// ---------------------------------------------------------------------------
// Pass 1: per-batch sum / sumsq of v.  256 blocks per batch so every block's
// data belongs to exactly one batch; wave32 shuffle tree + LDS + f32 atomics.
// ---------------------------------------------------------------------------
__global__ __launch_bounds__(256) void bgu_reduce(const float* __restrict__ x,
                                                  float* __restrict__ ws)
{
    const int b = blockIdx.x >> 8;           // 256 blocks per batch
    const int j = blockIdx.x & 255;
    const v4f* v = (const v4f*)(x + (size_t)b * XB + VB);
    const size_t base = (size_t)j * 8192;    // 8192 float4 per block

    float s = 0.f, q = 0.f;
    for (int i = threadIdx.x; i < 8192; i += 256) {
        v4f f = v[base + i];
        s += f.x + f.y + f.z + f.w;
        q += f.x*f.x + f.y*f.y + f.z*f.z + f.w*f.w;
    }
    #pragma unroll
    for (int off = 16; off; off >>= 1) {
        s += __shfl_down(s, off, 32);
        q += __shfl_down(q, off, 32);
    }
    __shared__ float ls[8], lq[8];
    const int wid = threadIdx.x >> 5, lane = threadIdx.x & 31;
    if (lane == 0) { ls[wid] = s; lq[wid] = q; }
    __syncthreads();
    if (threadIdx.x == 0) {
        float ts = 0.f, tq = 0.f;
        #pragma unroll
        for (int w = 0; w < 8; ++w) { ts += ls[w]; tq += lq[w]; }
        atomicAdd(&ws[2*b],     ts);
        atomicAdd(&ws[2*b + 1], tq);
    }
}

// ---------------------------------------------------------------------------
// Pass 2a: finalize mean / inv_std per batch (8 values).
// ---------------------------------------------------------------------------
__global__ void bgu_finalize(float* __restrict__ ws)
{
    const int b = threadIdx.x;
    if (b < 8) {
        const float invN = 1.0f / (float)VB;
        const float m    = ws[2*b] * invN;
        const float var  = ws[2*b + 1] * invN - m * m;
        ws[16 + 2*b] = m;
        ws[17 + 2*b] = rsqrtf(var + EPS);
    }
}

// ---------------------------------------------------------------------------
// Pass 2b: pre-split W into f16 hi/lo B-fragments in the exact wave32 WMMA
// B layout so the GEMM kernel does zero conversion VALU:
//   fragment f = ntile*8 + kstep ;  lane holds 16 halves:
//   halves[0..7]  = W[n][kb+0..7],  halves[8..15] = W[n][kb+16..23]
//   with n = ntile*16 + (lane&15),  kb = kstep*32 + 8*(lane>=16).
// ---------------------------------------------------------------------------
__global__ __launch_bounds__(32) void bgu_prepW(const float* __restrict__ Wm,
                                                _Float16* __restrict__ bhi,
                                                _Float16* __restrict__ blo)
{
    const int f    = blockIdx.x;             // 0..127
    const int nt   = f >> 3;
    const int kk   = f & 7;
    const int lane = threadIdx.x;            // 0..31
    const int khalf = (lane >> 4) << 3;
    const float* wrow = Wm + (size_t)(nt*16 + (lane & 15)) * 256 + kk*32 + khalf;

    v16h hi, lo;
    #pragma unroll
    for (int e = 0; e < 8; ++e) {
        const float w0 = wrow[e];
        const float w1 = wrow[16 + e];
        const _Float16 h0 = (_Float16)w0;
        const _Float16 h1 = (_Float16)w1;
        hi[e]     = h0;  lo[e]     = (_Float16)(w0 - (float)h0);
        hi[8 + e] = h1;  lo[8 + e] = (_Float16)(w1 - (float)h1);
    }
    const size_t off = ((size_t)f * 32 + lane) * 16;     // halves
    *(v16h*)(bhi + off) = hi;
    *(v16h*)(blo + off) = lo;
}

// ---------------------------------------------------------------------------
// Pass 3: fused  normalize -> (v_norm @ W^T + bias) -> u*(y+1).
// Block = 256 threads = 8 waves covering 32 rows x 256 cols; wave (w&3) owns
// cols [64w, 64w+64) as four 16x16 tiles, waves 0-3 rows [0,16), 4-7 [16,32).
// K = 256 in 8 chunks of 32; fp32-quality via f16 hi/lo split (3 WMMAs).
// All 8 B fragments are loaded up-front each K-step (one long load clause)
// so WMMAs drain the loads with staged partial waits instead of wait(0).
// ---------------------------------------------------------------------------
__global__ __launch_bounds__(256) void bgu_main(const float* __restrict__ x,
                                                const float* __restrict__ bias,
                                                float* __restrict__ out,
                                                const float* __restrict__ ws,
                                                const v16h* __restrict__ bh,
                                                const v16h* __restrict__ bl)
{
    const int lane  = threadIdx.x & 31;
    const int wid   = threadIdx.x >> 5;
    const int mtile = wid >> 2;              // 0 or 1
    const int nq    = wid & 3;               // col quarter
    const int n0    = nq * 64;

    const int R0   = blockIdx.x * 32 + mtile * 16;  // first global v-row
    const int b    = R0 >> 15;                      // 32768 rows per batch
    const int rib0 = R0 & 32767;                    // row-in-batch

    const float inv = ws[17 + 2*b];
    const float nmi = -ws[16 + 2*b] * inv;          // fold: an = a*inv + nmi

    const float* vrow = x + (size_t)b * XB + VB
                          + (size_t)(rib0 + (lane & 15)) * 256;
    const int khalf = (lane >> 4) << 3;      // 0 or 8

    v8f acc[4] = {};

    for (int kk = 0; kk < 8; ++kk) {
        const int kb = kk * 32 + khalf;

        // ---- issue ALL loads for this K-step first: A (4x b128) + B (16x b128)
        const float* ap = vrow + kb;
        v4f f0 = *(const v4f*)(ap);
        v4f f1 = *(const v4f*)(ap + 4);
        v4f f2 = *(const v4f*)(ap + 16);
        v4f f3 = *(const v4f*)(ap + 20);

        v16h bhf[4], blf[4];
        #pragma unroll
        for (int t = 0; t < 4; ++t) {
            const size_t fi = ((size_t)((nq*4 + t)*8 + kk)) * 32 + lane;
            bhf[t] = bh[fi];
            blf[t] = bl[fi];
        }

        // ---- normalize + hi/lo split A fragment (overlaps B loads) ----
        float af[16] = { f0.x,f0.y,f0.z,f0.w, f1.x,f1.y,f1.z,f1.w,
                         f2.x,f2.y,f2.z,f2.w, f3.x,f3.y,f3.z,f3.w };
        v16h a_hi, a_lo;
        #pragma unroll
        for (int e = 0; e < 16; ++e) {
            const float an = fmaf(af[e], inv, nmi);
            const _Float16 h = (_Float16)an;
            a_hi[e] = h;
            a_lo[e] = (_Float16)(an - (float)h);
        }

        // ---- 12 WMMAs drain the load clause tile by tile ----
        #pragma unroll
        for (int t = 0; t < 4; ++t) {
            acc[t] = __builtin_amdgcn_wmma_f32_16x16x32_f16(
                         false, a_hi, false, bhf[t], (short)0, acc[t], false, false);
            acc[t] = __builtin_amdgcn_wmma_f32_16x16x32_f16(
                         false, a_lo, false, bhf[t], (short)0, acc[t], false, false);
            acc[t] = __builtin_amdgcn_wmma_f32_16x16x32_f16(
                         false, a_hi, false, blf[t], (short)0, acc[t], false, false);
        }
    }

    // ---- fused epilogue: out = u * (y + bias + 1) ----
    // D layout: VGPR r -> M = 8*(lane>=16) + r ; N = n0 + t*16 + (lane&15)
    // u / out are touched exactly once -> non-temporal so the streaming
    // 512 MB doesn't evict the hot W-fragment table from L2.
    const float* ub = x   + (size_t)b * XB;
    float*       ob = out + (size_t)b * VB;
    const int mbase = (lane >> 4) * 8;
    #pragma unroll
    for (int t = 0; t < 4; ++t) {
        const int n    = n0 + t*16 + (lane & 15);
        const float bn = bias[n] + 1.0f;
        #pragma unroll
        for (int r = 0; r < 8; ++r) {
            const size_t off = (size_t)(rib0 + mbase + r) * 256 + n;
            const float uu = __builtin_nontemporal_load(ub + off);
            __builtin_nontemporal_store(uu * (acc[t][r] + bn), ob + off);
        }
    }
}

// ---------------------------------------------------------------------------
extern "C" void kernel_launch(void* const* d_in, const int* in_sizes, int n_in,
                              void* d_out, int out_size, void* d_ws, size_t ws_size,
                              hipStream_t stream)
{
    (void)in_sizes; (void)n_in; (void)out_size; (void)ws_size;
    const float* x    = (const float*)d_in[0];
    const float* Wm   = (const float*)d_in[1];
    const float* bias = (const float*)d_in[2];
    float* out = (float*)d_out;
    float* ws  = (float*)d_ws;
    _Float16* bhi = (_Float16*)((char*)d_ws + BHI_OFF);
    _Float16* blo = (_Float16*)((char*)d_ws + BLO_OFF);

    // ws[0..15] = per-batch {sum,sumsq}; ws[16..31] = {mean, inv_std};
    // then 2 x 128KB of pre-split W fragments.
    hipMemsetAsync(d_ws, 0, 32 * sizeof(float), stream);
    bgu_prepW   <<<128,   32, 0, stream>>>(Wm, bhi, blo);
    bgu_reduce  <<<2048, 256, 0, stream>>>(x, ws);
    bgu_finalize<<<1,      8, 0, stream>>>(ws);
    bgu_main    <<<8192, 256, 0, stream>>>(x, bias, out, ws,
                                           (const v16h*)bhi, (const v16h*)blo);
}
